// Conv2dBin_27994596835419
// MI455X (gfx1250) — matrix operand order repaired and checked by reading, not compiled, not run on hardware
//
#include <hip/hip_runtime.h>

// ---- problem constants (from reference) ----
#define N_IMG  32
#define C_IN   256
#define HDIM   56
#define WDIM   56
#define HW     3136      // 56*56
#define K_OUT  256
#define CRS    2304      // 256*3*3
// ---- tiling ----
#define MT     128       // out-channel block tile
#define NT     256       // spatial block tile (13 tiles/image, last is partial)
#define KB     32        // K-step per WMMA (one channel chunk)
#define NTILES 13        // ceil(3136/256)

// CDNA5 async global->LDS path (guarded so a missing builtin cannot break compile)
#if defined(__has_builtin)
#  if __has_builtin(__builtin_amdgcn_global_load_async_to_lds_b128) && \
      __has_builtin(__builtin_amdgcn_s_wait_asynccnt)
#    define USE_ASYNC_LDS 1
#  endif
#endif
#ifndef USE_ASYNC_LDS
#  define USE_ASYNC_LDS 0
#endif

typedef __attribute__((ext_vector_type(16))) __bf16 v16bf;
typedef __attribute__((ext_vector_type(8)))  __bf16 v8bf;
typedef __attribute__((ext_vector_type(8)))  float  v8f;
typedef __attribute__((ext_vector_type(4)))  int    v4i_t;

__device__ __forceinline__ __bf16 sign_bf16(float v) {
    unsigned short b = (v > 0.0f) ? (unsigned short)0x3F80
                     : (v < 0.0f) ? (unsigned short)0xBF80
                                  : (unsigned short)0x0000;
    return __builtin_bit_cast(__bf16, b);
}

// One-time: binarize weights into [tap][k_out][c] bf16 layout so the GEMM
// A-tile staging is a pure async DMA copy (no compares/selects in the hot loop).
__global__ __launch_bounds__(256)
void prepack_kernel(const float* __restrict__ w, __bf16* __restrict__ wp) {
    const int o   = blockIdx.x * 256 + threadIdx.x;  // 9*256*256 elements
    const int c   = o & 255;
    const int tm  = o >> 8;            // tap*256 + m
    const int m   = tm & 255;
    const int tap = tm >> 8;
    wp[o] = sign_bf16(w[(size_t)m * CRS + c * 9 + tap]);
}

template <bool PRE>
__global__ __launch_bounds__(256)
void Conv2dBin_wmma_kernel(const float* __restrict__ x,
                           const float* __restrict__ wraw,
                           const __bf16* __restrict__ wpack,
                           const float* __restrict__ bias,
                           float* __restrict__ out) {
    __shared__ alignas(32) __bf16 lA[MT * KB];  // 8 KB  [m][k]
    __shared__ alignas(32) __bf16 lB[NT * KB];  // 16 KB [n][k]

    const int tid   = threadIdx.x;
    const int wave  = tid >> 5;
    const int lane  = tid & 31;
    const int waveM = wave & 1;    // 2 waves across M (64 rows each)
    const int waveN = wave >> 1;   // 4 waves across N (64 cols each)
    const int lm    = lane & 15;
    const int kh    = lane >> 4;

    const int tileIdx = blockIdx.x;          // 0..415
    const int img     = tileIdx / NTILES;
    const int pos0    = (tileIdx % NTILES) * NT;
    const int m0      = blockIdx.y * MT;

    const float* xImg = x + (size_t)img * C_IN * HW;

    // Per-thread gather coordinates: thread tid owns spatial column n = tid.
    const int  pos   = pos0 + tid;
    const bool posOK = pos < HW;
    const int  oh    = pos / WDIM;
    const int  ow    = pos - oh * WDIM;

    // A-staging coordinates
    const int am    = tid >> 1;
    const int ahalf = tid & 1;

    v8f acc[4][4] = {};

    for (int tap = 0; tap < 9; ++tap) {
        const int r = tap / 3;
        const int s = tap - r * 3;
        // Tap-constant gather predicate, masked-zero bits, clamped base pointer
        const int  ih  = oh + r - 1;
        const int  iw  = ow + s - 1;
        const bool inb = posOK && (unsigned)ih < (unsigned)HDIM
                               && (unsigned)iw < (unsigned)WDIM;
        const unsigned short msk = inb ? (unsigned short)0xFFFFu : (unsigned short)0u;
        // Always-dereferenceable base: OOB lanes read xImg[0..] and mask to zero.
        const float* bsafe = inb ? (xImg + ih * WDIM + iw) : xImg;

        for (int c0 = 0; c0 < C_IN; c0 += KB) {
            __syncthreads();

            // ---- stage A: binarized weights, [m][c] contiguous ----
            {
                if (PRE) {
                    const __bf16* src = wpack
                        + ((size_t)(tap * K_OUT + m0 + am)) * C_IN + c0 + ahalf * 16;
                    __bf16* dst = &lA[am * KB + ahalf * 16];
#if USE_ASYNC_LDS
                    // CDNA5 async DMA: global -> LDS, tracked by ASYNCcnt
                    auto g0 = (__attribute__((address_space(1))) v4i_t*)(uintptr_t)src;
                    auto l0 = (__attribute__((address_space(3))) v4i_t*)
                                  (unsigned)(uintptr_t)dst;
                    __builtin_amdgcn_global_load_async_to_lds_b128(g0,     l0,     0, 0);
                    __builtin_amdgcn_global_load_async_to_lds_b128(g0 + 1, l0 + 1, 0, 0);
#else
                    ((uint4*)dst)[0] = ((const uint4*)src)[0];
                    ((uint4*)dst)[1] = ((const uint4*)src)[1];
#endif
                } else {
                    const float* src = wraw + (size_t)(m0 + am) * CRS
                                            + (size_t)(c0 + ahalf * 16) * 9 + tap;
                    __bf16* dst = &lA[am * KB + ahalf * 16];
                    #pragma unroll
                    for (int j = 0; j < 16; ++j)
                        dst[j] = sign_bf16(src[(size_t)j * 9]);
                }
            }

            // ---- stage B: branchless stride-HW channel loads ----
            {
                alignas(16) unsigned short tmp[KB];
                #pragma unroll
                for (int j = 0; j < KB; ++j) {
                    const float v = bsafe[(size_t)(c0 + j) * HW];
                    const unsigned short h =
                        __builtin_bit_cast(unsigned short, (__bf16)v);
                    tmp[j] = (unsigned short)(h & msk);
                }
                uint4* dst = (uint4*)&lB[tid * KB];
                const uint4* st = (const uint4*)tmp;
                dst[0] = st[0]; dst[1] = st[1]; dst[2] = st[2]; dst[3] = st[3];
            }

#if USE_ASYNC_LDS
            if (PRE) __builtin_amdgcn_s_wait_asynccnt(0);
#endif
            __syncthreads();

            // ---- fragments (A 16x32 / B 32x16, both in contiguous k = 16*kh + s order) ----
            v16bf bfrag[4];
            #pragma unroll
            for (int nt = 0; nt < 4; ++nt)
                bfrag[nt] = *(const v16bf*)
                    &lB[(waveN * 64 + nt * 16 + lm) * KB + kh * 16];

            #pragma unroll
            for (int mt = 0; mt < 4; ++mt) {
                const v8bf* rp = (const v8bf*)&lA[(waveM * 64 + mt * 16 + lm) * KB];
                const v8bf lo = rp[2 * kh];
                const v8bf hi = rp[2 * kh + 1];
                const v16bf a = __builtin_shufflevector(lo, hi,
                    0,1,2,3,4,5,6,7,8,9,10,11,12,13,14,15);
                #pragma unroll
                for (int nt = 0; nt < 4; ++nt)
                    acc[mt][nt] = __builtin_amdgcn_wmma_f32_16x16x32_bf16(
                        false, a, false, bfrag[nt],
                        (short)0, acc[mt][nt], false, false);
            }
        }
    }

    // ---- epilogue: C/D layout (VGPR d -> M = kh*8 + d), fused bias ----
    const int mrb = kh * 8;
    #pragma unroll
    for (int mt = 0; mt < 4; ++mt) {
        #pragma unroll
        for (int nt = 0; nt < 4; ++nt) {
            const int nGlob = pos0 + waveN * 64 + nt * 16 + lm;
            if (nGlob < HW) {
                #pragma unroll
                for (int d = 0; d < 8; ++d) {
                    const int mGlob = m0 + waveM * 64 + mt * 16 + mrb + d;
                    out[((size_t)img * K_OUT + mGlob) * HW + nGlob] =
                        acc[mt][nt][d] + bias[mGlob];
                }
            }
        }
    }
}

extern "C" void kernel_launch(void* const* d_in, const int* in_sizes, int n_in,
                              void* d_out, int out_size, void* d_ws, size_t ws_size,
                              hipStream_t stream) {
    const float* x  = (const float*)d_in[0];  // (32,256,56,56)
    const float* w  = (const float*)d_in[1];  // (256,256,3,3)
    const float* b  = (const float*)d_in[2];  // (256,)
    float* out = (float*)d_out;               // (32,256,56,56)

    constexpr size_t WS_NEED = (size_t)9 * K_OUT * C_IN * sizeof(unsigned short);
    dim3 grid(N_IMG * NTILES, K_OUT / MT);    // (416, 2)

    if (ws_size >= WS_NEED) {
        __bf16* wp = (__bf16*)d_ws;
        prepack_kernel<<<9 * K_OUT * C_IN / 256, 256, 0, stream>>>(w, wp);
        Conv2dBin_wmma_kernel<true><<<grid, 256, 0, stream>>>(x, w, wp, b, out);
    } else {
        Conv2dBin_wmma_kernel<false><<<grid, 256, 0, stream>>>(x, w, nullptr, b, out);
    }
}
